// FP8Linear_64604898066760
// MI455X (gfx1250) — 8-bit matrix operand order repaired and checked by reading, compiled, not run on hardware
//
#include <hip/hip_runtime.h>


typedef __attribute__((ext_vector_type(16))) int   v16i;
typedef __attribute__((ext_vector_type(8)))  float v8f;
typedef int v2i_ __attribute__((vector_size(8)));
typedef int v4i_ __attribute__((vector_size(16)));

#define AS1 __attribute__((address_space(1)))
#define AS3 __attribute__((address_space(3)))

#define FP8_MAX 448.0f
#define QEPS 1e-12f

#if __has_builtin(__builtin_amdgcn_global_load_async_to_lds_b64) && \
    __has_builtin(__builtin_amdgcn_global_load_async_to_lds_b128)
#define HAVE_ASYNC_LDS 1
#else
#define HAVE_ASYNC_LDS 0
#endif

__device__ inline void wait_async0() {
#if HAVE_ASYNC_LDS
#if __has_builtin(__builtin_amdgcn_s_wait_asynccnt)
    __builtin_amdgcn_s_wait_asynccnt(0);
#else
    asm volatile("s_wait_asynccnt 0x0" ::: "memory");
#endif
#endif
}

__device__ inline unsigned char sw_f32_to_e4m3(float f) {
    unsigned u = __float_as_uint(f);
    unsigned s = (u >> 31) << 7;
    float a = fabsf(f);
    if (!(a > 0.0f)) return (unsigned char)s;
    if (a >= 464.0f) return (unsigned char)(s | 0x7E);
    int ex;
    (void)frexpf(a, &ex);
    int qe = ex - 1;
    if (qe < -6) qe = -6;
    float sc = exp2f((float)(3 - qe));
    int mant = (int)rintf(a * sc);
    if (mant >= 16) { mant = 8; qe++; }
    int bits = (mant < 8) ? mant : (((qe + 7) << 3) | (mant - 8));
    if (bits > 0x7E) bits = 0x7E;
    return (unsigned char)(s | bits);
}

__device__ inline int pack4_e4m3(float a, float b, float c, float d) {
#if __has_builtin(__builtin_amdgcn_cvt_pk_fp8_f32)
    int r = 0;
    r = __builtin_amdgcn_cvt_pk_fp8_f32(a, b, r, false);
    r = __builtin_amdgcn_cvt_pk_fp8_f32(c, d, r, true);
    return r;
#else
    return (int)sw_f32_to_e4m3(a)        |
           ((int)sw_f32_to_e4m3(b) << 8) |
           ((int)sw_f32_to_e4m3(c) << 16)|
           ((int)sw_f32_to_e4m3(d) << 24);
#endif
}

__global__ void init_slots(unsigned* slots) {
    if (threadIdx.x < 2) slots[threadIdx.x] = 0u;
}

__global__ __launch_bounds__(256) void absmax_kernel(const float* __restrict__ src,
                                                     long long n4, unsigned* slot) {
    long long i = (long long)blockIdx.x * blockDim.x + threadIdx.x;
    long long stride = (long long)gridDim.x * blockDim.x;
    float m = 0.0f;
    for (; i < n4; i += stride) {
        float4 v = ((const float4*)src)[i];
        m = fmaxf(m, fmaxf(fmaxf(fabsf(v.x), fabsf(v.y)),
                           fmaxf(fabsf(v.z), fabsf(v.w))));
    }
    #pragma unroll
    for (int off = 16; off > 0; off >>= 1)
        m = fmaxf(m, __shfl_xor(m, off, 32));
    if ((threadIdx.x & 31) == 0)
        atomicMax(slot, __float_as_uint(m));
}

__global__ __launch_bounds__(256) void quant_kernel(const float* __restrict__ src,
                                                    unsigned char* __restrict__ dst,
                                                    long long n4,
                                                    const unsigned* __restrict__ slot) {
    const float amax = __uint_as_float(slot[0]);
    const float s = FP8_MAX / (amax + QEPS);
    long long i = (long long)blockIdx.x * blockDim.x + threadIdx.x;
    long long stride = (long long)gridDim.x * blockDim.x;
    for (; i < n4; i += stride) {
        float4 v = ((const float4*)src)[i];
        ((int*)dst)[i] = pack4_e4m3(v.x * s, v.y * s, v.z * s, v.w * s);
    }
}

__device__ inline void stage_tile(AS3 unsigned char* buf,
                                  const AS1 unsigned char* __restrict__ A,
                                  const AS1 unsigned char* __restrict__ B,
                                  int m0, int n0, int kk, int K, int tid)
{
#if HAVE_ASYNC_LDS
    #pragma unroll
    for (int i = 0; i < 8; ++i) {
        int t = tid + i * 256;
        int m = t >> 4, k0 = (t & 15) << 3;
        int mt = m >> 4, ml = m & 15, j = ((k0 >> 5) << 1) | ((k0 >> 3) & 1), hi = (k0 & 16) ? 16 : 0;
        __builtin_amdgcn_global_load_async_to_lds_b64(
            (AS1 v2i_*)(A + (size_t)(m0 + m) * K + kk + k0),
            (AS3 v2i_*)(buf + mt * 2048 + (ml + hi) * 64 + j * 8), 0, 0);
    }
    #pragma unroll
    for (int i = 0; i < 4; ++i) {
        int t = tid + i * 256;
        int n = t >> 3, k0 = (t & 7) << 4;
        int nt = n >> 4, nl = n & 15, lp = (k0 & 16) ? (16 + nl) : nl;
        __builtin_amdgcn_global_load_async_to_lds_b128(
            (AS1 v4i_*)(B + (size_t)(n0 + n) * K + kk + k0),
            (AS3 v4i_*)(buf + 16384 + nt * 2048 + lp * 64 + (k0 >> 5) * 16), 0, 0);
    }
#else
    #pragma unroll
    for (int i = 0; i < 8; ++i) {
        int t = tid + i * 256;
        int m = t >> 4, k0 = (t & 15) << 3;
        unsigned long long d =
            *(const AS1 unsigned long long*)(A + (size_t)(m0 + m) * K + kk + k0);
        int mt = m >> 4, ml = m & 15, j = ((k0 >> 5) << 1) | ((k0 >> 3) & 1), hi = (k0 & 16) ? 16 : 0;
        *(AS3 unsigned long long*)(buf + mt * 2048 + (ml + hi) * 64 + j * 8) = d;
    }
    #pragma unroll
    for (int i = 0; i < 4; ++i) {
        int t = tid + i * 256;
        int n = t >> 3, k0 = (t & 7) << 4;
        v4i_ d = *(const AS1 v4i_*)(B + (size_t)(n0 + n) * K + kk + k0);
        int nt = n >> 4, nl = n & 15, lp = (k0 & 16) ? (16 + nl) : nl;
        *(AS3 v4i_*)(buf + 16384 + nt * 2048 + lp * 64 + (k0 >> 5) * 16) = d;
    }
#endif
}

__global__ __launch_bounds__(256) void gemm_fp8_wmma(
    const unsigned char* __restrict__ Ag,
    const unsigned char* __restrict__ Bg,
    const float* __restrict__ bias,
    const unsigned* __restrict__ slots,
    float* __restrict__ out,
    int M, int N, int K)
{
    __shared__ __attribute__((aligned(64))) unsigned char lds[2][32768];

    const int tid  = threadIdx.x;
    const int lane = tid & 31;
    const int wave = tid >> 5;
    const int m0 = blockIdx.y * 128;
    const int n0 = blockIdx.x * 128;
    const int wm = (wave & 1) * 64;
    const int wn = (wave >> 1) * 32;

    AS3 unsigned char* lbase = (AS3 unsigned char*)lds;
    const AS1 unsigned char* A = (const AS1 unsigned char*)Ag;
    const AS1 unsigned char* B = (const AS1 unsigned char*)Bg;

    const float ax = __uint_as_float(slots[0]);
    const float aw = __uint_as_float(slots[1]);
    const float oscale = (ax + QEPS) * (aw + QEPS) * (1.0f / (FP8_MAX * FP8_MAX));

    v8f acc[4][2] = {};

    const int T = K >> 7;
    stage_tile(lbase, A, B, m0, n0, 0, K, tid);

    for (int kt = 0; kt < T; ++kt) {
        wait_async0();
        __syncthreads();

        if (kt + 1 < T)
            stage_tile(lbase + ((kt + 1) & 1) * 32768, A, B,
                       m0, n0, (kt + 1) << 7, K, tid);

        AS3 unsigned char* cur = lbase + (kt & 1) * 32768;

        v16i afrag[4], bfrag[2];
        #pragma unroll
        for (int mi = 0; mi < 4; ++mi)
            afrag[mi] = *(const AS3 v16i*)(cur + ((wm >> 4) + mi) * 2048 + lane * 64);
        #pragma unroll
        for (int ni = 0; ni < 2; ++ni)
            bfrag[ni] = *(const AS3 v16i*)(cur + 16384 + ((wn >> 4) + ni) * 2048 + lane * 64);

        #pragma unroll
        for (int mi = 0; mi < 4; ++mi)
            #pragma unroll
            for (int ni = 0; ni < 2; ++ni)
                acc[mi][ni] = __builtin_amdgcn_wmma_f32_16x16x128_fp8_fp8(
                    afrag[mi], bfrag[ni], (short)0, acc[mi][ni], false, false);
    }

    #pragma unroll
    for (int mi = 0; mi < 4; ++mi) {
        int mrow = m0 + wm + mi * 16 + ((lane >= 16) ? 8 : 0);
        #pragma unroll
        for (int ni = 0; ni < 2; ++ni) {
            int col = n0 + wn + ni * 16 + (lane & 15);
            float bv = bias[col];
            #pragma unroll
            for (int e = 0; e < 8; ++e)
                out[(size_t)(mrow + e) * N + col] = acc[mi][ni][e] * oscale + bv;
        }
    }
}

extern "C" void kernel_launch(void* const* d_in, const int* in_sizes, int n_in,
                              void* d_out, int out_size, void* d_ws, size_t ws_size,
                              hipStream_t stream) {
    const float* x    = (const float*)d_in[0];
    const float* w    = (const float*)d_in[1];
    const float* bias = (const float*)d_in[2];

    const long long xcount = in_sizes[0];
    const long long wcount = in_sizes[1];
    const int N = in_sizes[2];
    const int K = (int)(wcount / N);
    const int M = (int)(xcount / K);

    unsigned*      slots = (unsigned*)d_ws;
    unsigned char* Aq    = (unsigned char*)d_ws + 256;
    unsigned char* Bq    = Aq + (size_t)xcount;
    float*         out   = (float*)d_out;

    init_slots<<<1, 64, 0, stream>>>(slots);
    absmax_kernel<<<1024, 256, 0, stream>>>(x, xcount / 4, slots + 0);
    absmax_kernel<<<512,  256, 0, stream>>>(w, wcount / 4, slots + 1);
    quant_kernel<<<2048, 256, 0, stream>>>(x, Aq, xcount / 4, slots + 0);
    quant_kernel<<<1024, 256, 0, stream>>>(w, Bq, wcount / 4, slots + 1);

    dim3 grid(N / 128, M / 128);
    gemm_fp8_wmma<<<grid, 256, 0, stream>>>(Aq, Bq, bias, slots, out, M, N, K);
}
